// DEC_80212809220613
// MI455X (gfx1250) — matrix operand order repaired and checked by reading, not compiled, not run on hardware
//
#include <hip/hip_runtime.h>

// ---------------------------------------------------------------------------
// Problem constants (from reference)
// ---------------------------------------------------------------------------
#define BB   128     // batch
#define TT   64      // time steps
#define HH   100     // hidden
#define NIN  3       // input features
#define DSH  10      // readout shift
#define G3   300     // 3*H (gate concat)

// WMMA tiling
#define BT      16   // batch tile per workgroup (WMMA M)
#define KP      128  // padded K for GRU GEMMs (H=100 -> 128)
#define NP      304  // padded N for GRU GEMMs (300 -> 304)
#define NT_GRU  19   // N tiles (304/16)
#define KI_GRU  4    // K iters (128/32)
#define KP2     256  // padded K for ctx GEMM (concat 2x128)
#define NT_CTX  7    // N tiles for ctx (112/16)
#define KI_CTX  8    // K iters for ctx (256/32)

#define FRAG_GRU (NT_GRU * KI_GRU * 32 * 16)   // 38912 halfs per GRU weight
#define FRAG_CTX (NT_CTX * KI_CTX * 32 * 16)   // 28672 halfs for ctx weight

typedef __bf16 bf16_t;
typedef bf16_t bf16x16 __attribute__((ext_vector_type(16)));
typedef bf16_t bf16x8  __attribute__((ext_vector_type(8)));
typedef float  f32x8   __attribute__((ext_vector_type(8)));

// ---------------------------------------------------------------------------
// LDS layout (dynamic shared memory, byte offsets, 16B aligned)
// ---------------------------------------------------------------------------
#define OFF_H    0               // float [2][BT][KP]   16384
#define OFF_HN   16384           // float [2][BT][KP]   16384
#define OFF_GI   32768           // float [BT][NP]      19456
#define OFF_GH   52224           // float [BT][NP]      19456
#define OFF_CTXA 71680           // bf16  [2][BT*KP2]   16384
#define OFF_HA0  88064           // bf16  [BT*KP]        4096
#define OFF_HA1  92160           // bf16  [BT*KP]        4096
#define OFF_CDOT 96256           // float [2][BT][TT]    8192
#define OFF_EBUF 104448          // float [2][BT][TT]    8192
#define OFF_DHN  112640          // float [2][BT]         128
#define OFF_XS   112768          // float [BT][NIN]       256
#define SMEM_TOTAL 113024

// ---------------------------------------------------------------------------
// Weight pre-swizzle kernels: pack bf16 WMMA B-fragments
// B-fragment (32x16 K x N): lane l, halfword j -> k = kIter*32 + j + 16*(l>>4),
//                                                 n = ntile*16 + (l&15)
// ---------------------------------------------------------------------------
__global__ void prep_gru_w(const float* __restrict__ W /* [300][100] */,
                           bf16_t* __restrict__ dst) {
  int idx = blockIdx.x * blockDim.x + threadIdx.x;
  if (idx >= FRAG_GRU) return;
  int j     = idx & 15;
  int lane  = (idx >> 4) & 31;
  int kIter = (idx >> 9) & 3;
  int ntile = idx >> 11;
  int k = kIter * 32 + j + 16 * (lane >> 4);
  int n = ntile * 16 + (lane & 15);
  float v = 0.f;
  if (n < G3 && k < HH) v = W[n * HH + k];
  dst[idx] = (bf16_t)v;
}

__global__ void prep_ctx_w(const float* __restrict__ W /* [100][200] */,
                           bf16_t* __restrict__ dst) {
  int idx = blockIdx.x * blockDim.x + threadIdx.x;
  if (idx >= FRAG_CTX) return;
  int j     = idx & 15;
  int lane  = (idx >> 4) & 31;
  int kIter = (idx >> 9) & 7;
  int ntile = idx >> 12;
  int k = kIter * 32 + j + 16 * (lane >> 4);   // [0,256): [0,128)=c part, [128,256)=hn part
  int n = ntile * 16 + (lane & 15);            // [0,112)
  float v = 0.f;
  if (n < HH) {
    if (k < KP) { if (k < HH) v = W[n * (2 * HH) + k]; }
    else        { int kk = k - KP; if (kk < HH) v = W[n * (2 * HH) + HH + kk]; }
  }
  dst[idx] = (bf16_t)v;
}

// ---------------------------------------------------------------------------
// Fragment loaders + WMMA tile helpers
// A-fragment (16x32, bf16, from LDS row-major [16][kp]):
//   lane l, halfword j -> m = l&15, k = kIter*32 + j + 16*(l>>4)
//   => two contiguous 8-half (16B) LDS loads per lane.
// ---------------------------------------------------------------------------
__device__ __forceinline__ bf16x16 load_a_frag(const bf16_t* base, int lane,
                                               int kIter, int kp) {
  int m = lane & 15, g = lane >> 4;
  const bf16_t* p = base + m * kp + kIter * 32 + 16 * g;
  bf16x8 lo = *(const bf16x8*)p;          // k = kbase + 0..7
  bf16x8 hi = *(const bf16x8*)(p + 8);    // k = kbase + 8..15
  bf16x16 r;
#pragma unroll
  for (int i = 0; i < 8; ++i) { r[i] = lo[i]; r[i + 8] = hi[i]; }
  return r;
}

// One 16x16 output tile of a GRU GEMM: D = A(16xKP) * Wfrag + bias(n)
__device__ __forceinline__ void gemm_tile_gru(const bf16_t* Abuf,
                                              const bf16_t* __restrict__ Wf,
                                              const float* __restrict__ bias,
                                              float* outBuf, int ntile, int lane) {
  int nl = lane & 15, g = lane >> 4;
  int n = ntile * 16 + nl;
  float bv = (n < G3) ? bias[n] : 0.f;
  f32x8 acc;
#pragma unroll
  for (int v = 0; v < 8; ++v) acc[v] = bv;   // bias broadcast over M via C operand
#pragma unroll
  for (int ki = 0; ki < KI_GRU; ++ki) {
    bf16x16 a = load_a_frag(Abuf, lane, ki, KP);
    const bf16_t* wp = Wf + ((size_t)(ntile * KI_GRU + ki) * 32 + lane) * 16;
    bf16x16 b = *(const bf16x16*)wp;        // 32B contiguous pre-swizzled fragment
    acc = __builtin_amdgcn_wmma_f32_16x16x32_bf16(false, a, false, b,
                                                  (short)0, acc, false, false);
  }
#pragma unroll
  for (int v = 0; v < 8; ++v) {
    int m = v + 8 * g;                       // C/D layout: vgpr v, lane group -> M
    outBuf[m * NP + n] = acc[v];
  }
}

// One 16x16 output tile of the ctx GEMM: h_layer = ctxA(16x256) * ctxWf + ctx_b
__device__ __forceinline__ void gemm_tile_ctx(const bf16_t* Abuf,
                                              const bf16_t* __restrict__ Wf,
                                              const float* __restrict__ ctx_b,
                                              float* hl, int ntile, int lane) {
  int nl = lane & 15, g = lane >> 4;
  int n = ntile * 16 + nl;                   // < 112
  float bv = (n < HH) ? ctx_b[n] : 0.f;
  f32x8 acc;
#pragma unroll
  for (int v = 0; v < 8; ++v) acc[v] = bv;
#pragma unroll
  for (int ki = 0; ki < KI_CTX; ++ki) {
    bf16x16 a = load_a_frag(Abuf, lane, ki, KP2);
    const bf16_t* wp = Wf + ((size_t)(ntile * KI_CTX + ki) * 32 + lane) * 16;
    bf16x16 b = *(const bf16x16*)wp;
    acc = __builtin_amdgcn_wmma_f32_16x16x32_bf16(false, a, false, b,
                                                  (short)0, acc, false, false);
  }
#pragma unroll
  for (int v = 0; v < 8; ++v) {
    int m = v + 8 * g;
    hl[m * KP + n] = acc[v];                 // cols >=100 get exact 0 (padded W/bias)
  }
}

__device__ __forceinline__ float sigmoid_(float x) {
  return 1.f / (1.f + __expf(-x));
}

// ---------------------------------------------------------------------------
// Persistent recurrence kernel: one WG = (batch tile of 16) x (one stack)
// ---------------------------------------------------------------------------
struct GruParams {
  const float* received;
  const float* ih0[2];  const float* bih0[2]; const float* bhh0[2];
  const float* bih1[2]; const float* bhh1[2];
  const float* attn_w;  const float* attn_b;  const float* ctx_b;
  const bf16_t* hh0f[2]; const bf16_t* ih1f[2]; const bf16_t* hh1f[2];
  const bf16_t* ctxwf;
  float* cache;   // [stack][layer][B][T][H] fp32
};

__global__ __launch_bounds__(256) void gru_attn_persistent(GruParams P) {
  extern __shared__ unsigned char smem[];
  float*  h    = (float*)(smem + OFF_H);     // [2][BT][KP]  mixed hidden (padded 0)
  float*  hn   = (float*)(smem + OFF_HN);    // [2][BT][KP]  raw GRU hidden
  float*  gi   = (float*)(smem + OFF_GI);    // [BT][NP]
  float*  gh   = (float*)(smem + OFF_GH);    // [BT][NP]
  bf16_t* ctxA = (bf16_t*)(smem + OFF_CTXA); // [2][BT*KP2]  concat(c, hn) bf16
  bf16_t* hA0  = (bf16_t*)(smem + OFF_HA0);  // [BT*KP] A staging
  bf16_t* hA1  = (bf16_t*)(smem + OFF_HA1);
  float*  cdot = (float*)(smem + OFF_CDOT);  // [2][BT][TT] incremental cache . attn_w2
  float*  ebuf = (float*)(smem + OFF_EBUF);  // [2][BT][TT] scores / softmax weights
  float*  dhn  = (float*)(smem + OFF_DHN);   // [2][BT]      hn . attn_w1
  float*  xs   = (float*)(smem + OFF_XS);    // [BT][NIN]

  const int tid  = threadIdx.x;
  const int lane = tid & 31;
  const int wave = tid >> 5;
  const int s    = blockIdx.x >> 3;          // stack 0/1
  const int b0   = (blockIdx.x & 7) * BT;    // batch base

  const float*  ih0  = P.ih0[s];
  const float*  bih0 = P.bih0[s];
  const float*  bhh0 = P.bhh0[s];
  const float*  bih1 = P.bih1[s];
  const float*  bhh1 = P.bhh1[s];
  const bf16_t* hh0f = P.hh0f[s];
  const bf16_t* ih1f = P.ih1f[s];
  const bf16_t* hh1f = P.hh1f[s];
  float* cacheS = P.cache + (size_t)s * 2 * BB * TT * HH;  // [layer][B][T][H]
  const float attn_b = P.attn_b[0];

  // init state + padded regions to zero
  for (int i = tid; i < 2 * BT * KP; i += 256) { h[i] = 0.f; hn[i] = 0.f; }
  for (int i = tid; i < 2 * BT * KP2; i += 256) ctxA[i] = (bf16_t)0.f;
  __syncthreads();

  for (int t = 0; t < TT; ++t) {
    // ---- stage x(t), build bf16 A matrices from current hidden state ----
    if (tid < BT * NIN) {
      int b = tid / NIN, k = tid % NIN;
      xs[b * NIN + k] = P.received[((size_t)(b0 + b) * TT + t) * NIN + k];
    }
    for (int i = tid; i < BT * KP; i += 256) {
      hA0[i] = (bf16_t)h[i];                 // layer 0 hidden
      hA1[i] = (bf16_t)h[BT * KP + i];       // layer 1 hidden
    }
    __syncthreads();

    // ---- layer 0: gi = x@Wih0^T + bih0 (K=3, VALU); gh = h0@Whh0^T + bhh0 (WMMA)
    for (int i = tid; i < BT * G3; i += 256) {
      int b = i / G3, n = i % G3;
      gi[b * NP + n] = bih0[n] + xs[b * NIN + 0] * ih0[n * NIN + 0]
                               + xs[b * NIN + 1] * ih0[n * NIN + 1]
                               + xs[b * NIN + 2] * ih0[n * NIN + 2];
    }
    for (int tile = wave; tile < NT_GRU; tile += 8)
      gemm_tile_gru(hA0, hh0f, bhh0, gh, tile, lane);
    __syncthreads();

    // ---- layer 0 pointwise gates -> hn[0]; refresh hA0 with h0_new ----
    for (int i = tid; i < BT * HH; i += 256) {
      int b = i / HH, hc = i % HH;
      float r  = sigmoid_(gi[b * NP + hc]          + gh[b * NP + hc]);
      float z  = sigmoid_(gi[b * NP + HH + hc]     + gh[b * NP + HH + hc]);
      float nn = tanhf  (gi[b * NP + 2 * HH + hc] + r * gh[b * NP + 2 * HH + hc]);
      float hv = (1.f - z) * nn + z * h[b * KP + hc];
      hn[b * KP + hc] = hv;
      hA0[b * KP + hc] = (bf16_t)hv;         // pads (>=100) keep previous zeros
    }
    __syncthreads();

    // ---- layer 1: gi = h0_new@Wih1^T + bih1 ; gh = h1@Whh1^T + bhh1 (both WMMA)
    for (int jb = wave; jb < 2 * NT_GRU; jb += 8) {
      if (jb < NT_GRU) gemm_tile_gru(hA0, ih1f, bih1, gi, jb, lane);
      else             gemm_tile_gru(hA1, hh1f, bhh1, gh, jb - NT_GRU, lane);
    }
    __syncthreads();

    // ---- layer 1 pointwise gates -> hn[1] ----
    for (int i = tid; i < BT * HH; i += 256) {
      int b = i / HH, hc = i % HH;
      float r  = sigmoid_(gi[b * NP + hc]          + gh[b * NP + hc]);
      float z  = sigmoid_(gi[b * NP + HH + hc]     + gh[b * NP + HH + hc]);
      float nn = tanhf  (gi[b * NP + 2 * HH + hc] + r * gh[b * NP + 2 * HH + hc]);
      hn[BT * KP + b * KP + hc] = (1.f - z) * nn + z * h[BT * KP + b * KP + hc];
    }
    __syncthreads();

    // ---- append raw hiddens to cache; incremental attention dot-products ----
    for (int i = tid; i < 2 * BT * HH; i += 256) {
      int l = i / (BT * HH), rem = i % (BT * HH);
      int b = rem / HH, hc = rem % HH;
      cacheS[(((size_t)l * BB + b0 + b) * TT + t) * HH + hc] = hn[l * BT * KP + b * KP + hc];
    }
    if (tid < 2 * BT) {
      int l = tid >> 4, b = tid & 15;
      float dh = 0.f, cd = 0.f;
      for (int hc = 0; hc < HH; ++hc) {
        float v = hn[l * BT * KP + b * KP + hc];
        dh += v * P.attn_w[hc];              // hn . attn_w[:H]
        cd += v * P.attn_w[HH + hc];         // hn . attn_w[H:]   (future cache row)
      }
      dhn[l * BT + b] = dh;
      cdot[(l * BT + b) * TT + t] = cd;
    }
    __threadfence_block();
    __syncthreads();

    if (t == 0) {  // attention result discarded at t==0: h <- hn
      for (int i = tid; i < 2 * BT * KP; i += 256) h[i] = hn[i];
      __syncthreads();
      continue;
    }

    // ---- scores e[l,b,t'] = hn.w1 + cache.w2 + b  (t' < t) ----
    for (int i = tid; i < 2 * BT * t; i += 256) {
      int l = i / (BT * t), rem = i % (BT * t);
      int b = rem / t, tp = rem % t;
      ebuf[(l * BT + b) * TT + tp] = dhn[l * BT + b] + cdot[(l * BT + b) * TT + tp] + attn_b;
    }
    __syncthreads();

    // ---- softmax over t' (32 rows, one thread each, t <= 64) ----
    if (tid < 2 * BT) {
      float* row = ebuf + tid * TT;
      float mx = -3.4e38f;
      for (int tp = 0; tp < t; ++tp) mx = fmaxf(mx, row[tp]);
      float sum = 0.f;
      for (int tp = 0; tp < t; ++tp) { float e = __expf(row[tp] - mx); row[tp] = e; sum += e; }
      float inv = 1.f / sum;
      for (int tp = 0; tp < t; ++tp) row[tp] *= inv;
    }
    __syncthreads();

    // ---- context c = a . cache (VALU) ; build bf16 concat(c, hn) A-matrix ----
    for (int i = tid; i < 2 * BT * HH; i += 256) {
      int l = i / (BT * HH), rem = i % (BT * HH);
      int b = rem / HH, hc = rem % HH;
      const float* cp = cacheS + (((size_t)l * BB + b0 + b) * TT) * HH + hc;
      const float* arow = ebuf + (l * BT + b) * TT;
      float acc = 0.f;
      for (int tp = 0; tp < t; ++tp) acc += arow[tp] * cp[(size_t)tp * HH];
      ctxA[l * BT * KP2 + b * KP2 + hc]      = (bf16_t)acc;                        // c part
      ctxA[l * BT * KP2 + b * KP2 + KP + hc] = (bf16_t)hn[l * BT * KP + b * KP + hc]; // hn part
    }
    __syncthreads();

    // ---- ctx projection (WMMA): h[l] = concat(c,hn) @ ctx_w^T + ctx_b ----
    for (int jb = wave; jb < 2 * NT_CTX; jb += 8) {
      int l = jb / NT_CTX, tile = jb % NT_CTX;
      gemm_tile_ctx(ctxA + l * BT * KP2, P.ctxwf, P.ctx_b, h + l * BT * KP, tile, lane);
    }
    __syncthreads();
  }
}

// ---------------------------------------------------------------------------
// Readout: out[b,i] = sigmoid( [o1[b,i], o2[b,min(i+D,T-1)]] . out_w + out_b )
// o1/o2 are exactly cache[stack][layer=1]
// ---------------------------------------------------------------------------
__global__ void readout_kernel(const float* __restrict__ cache,
                               const float* __restrict__ out_w,
                               const float* __restrict__ out_b,
                               float* __restrict__ out) {
  int idx = blockIdx.x * blockDim.x + threadIdx.x;
  if (idx >= BB * TT) return;
  int b = idx / TT, i = idx % TT;
  int i2 = i + DSH; if (i2 > TT - 1) i2 = TT - 1;
  const float* c1 = cache + (((size_t)(0 * 2 + 1) * BB + b) * TT + i)  * HH;
  const float* c2 = cache + (((size_t)(1 * 2 + 1) * BB + b) * TT + i2) * HH;
  float acc = out_b[0];
  for (int hc = 0; hc < HH; ++hc)
    acc += c1[hc] * out_w[hc] + c2[hc] * out_w[HH + hc];
  out[idx] = 1.f / (1.f + __expf(-acc));
}

// ---------------------------------------------------------------------------
// Launch
// ---------------------------------------------------------------------------
extern "C" void kernel_launch(void* const* d_in, const int* in_sizes, int n_in,
                              void* d_out, int out_size, void* d_ws, size_t ws_size,
                              hipStream_t stream) {
  (void)in_sizes; (void)n_in; (void)out_size; (void)ws_size;
  // input order: 0 received | w1: 1 ih0, 2 hh0, 3 bih0, 4 bhh0, 5 ih1, 6 hh1, 7 bih1, 8 bhh1
  //              w2: 9..16 same | 17 attn_w, 18 attn_b, 19 ctx_w, 20 ctx_b, 21 out_w, 22 out_b
  unsigned char* ws = (unsigned char*)d_ws;
  float* cache = (float*)ws;                                   // 2*2*B*T*H fp32 = 13.1 MB
  size_t off = (size_t)2 * 2 * BB * TT * HH * sizeof(float);
  bf16_t* wf[6];
  for (int i = 0; i < 6; ++i) { wf[i] = (bf16_t*)(ws + off); off += (size_t)FRAG_GRU * sizeof(bf16_t); }
  bf16_t* ctxwf = (bf16_t*)(ws + off); off += (size_t)FRAG_CTX * sizeof(bf16_t);

  // pre-swizzle weights into WMMA B-fragment order (bf16)
  int gblk = (FRAG_GRU + 255) / 256, cblk = (FRAG_CTX + 255) / 256;
  prep_gru_w<<<gblk, 256, 0, stream>>>((const float*)d_in[2],  wf[0]); // w1_hh0
  prep_gru_w<<<gblk, 256, 0, stream>>>((const float*)d_in[5],  wf[1]); // w1_ih1
  prep_gru_w<<<gblk, 256, 0, stream>>>((const float*)d_in[6],  wf[2]); // w1_hh1
  prep_gru_w<<<gblk, 256, 0, stream>>>((const float*)d_in[10], wf[3]); // w2_hh0
  prep_gru_w<<<gblk, 256, 0, stream>>>((const float*)d_in[13], wf[4]); // w2_ih1
  prep_gru_w<<<gblk, 256, 0, stream>>>((const float*)d_in[14], wf[5]); // w2_hh1
  prep_ctx_w<<<cblk, 256, 0, stream>>>((const float*)d_in[19], ctxwf); // ctx_w

  GruParams P;
  P.received = (const float*)d_in[0];
  P.ih0[0]  = (const float*)d_in[1];  P.bih0[0] = (const float*)d_in[3];
  P.bhh0[0] = (const float*)d_in[4];  P.bih1[0] = (const float*)d_in[7];
  P.bhh1[0] = (const float*)d_in[8];
  P.ih0[1]  = (const float*)d_in[9];  P.bih0[1] = (const float*)d_in[11];
  P.bhh0[1] = (const float*)d_in[12]; P.bih1[1] = (const float*)d_in[15];
  P.bhh1[1] = (const float*)d_in[16];
  P.attn_w = (const float*)d_in[17];  P.attn_b = (const float*)d_in[18];
  P.ctx_b  = (const float*)d_in[20];
  P.hh0f[0] = wf[0]; P.ih1f[0] = wf[1]; P.hh1f[0] = wf[2];
  P.hh0f[1] = wf[3]; P.ih1f[1] = wf[4]; P.hh1f[1] = wf[5];
  P.ctxwf = ctxwf;
  P.cache = cache;

  gru_attn_persistent<<<16, 256, SMEM_TOTAL, stream>>>(P);

  readout_kernel<<<(BB * TT + 255) / 256, 256, 0, stream>>>(
      cache, (const float*)d_in[21], (const float*)d_in[22], (float*)d_out);
}
